// HybridModel_52441550684718
// MI455X (gfx1250) — compile-verified, matrix operand order checked by reading, not run on hardware
//
#include <hip/hip_runtime.h>
#include <hip/hip_bf16.h>

typedef __attribute__((ext_vector_type(16))) _Float16 v16h;
typedef __attribute__((ext_vector_type(8)))  float    v8f;

// ---------------------------------------------------------------------------
// Kernel 1: fused FC1 (relu) + FC2 -> x_enc (4096 x 64, f32) via WMMA f16
//   h     = relu(x @ W1^T + b1)   x:(B,2048)  W1:(128,2048)
//   x_enc = h @ W2^T + b2         W2:(64,128)
// Block: 128 threads (4 waves), 64 rows of x per block, N=128 (all of W1).
// ---------------------------------------------------------------------------

#define KT 32          // K tile (matches wmma 16x16x32)
#define APAD 8         // LDS padding (halfs) to dodge bank conflicts
#define HPAD 8

// Fragment load from an LDS tile laid out [row][stride] (f16).
// 16-bit A/B fragment layout (CDNA5 ISA §7.12.2, 16-bit 16x32):
//   lanes 0-15 : row = lane,  elems j=0..7 -> K=j,    j=8..15 -> K=j+8
//   lanes 16-31: row = lane-16, same but K offset +8
__device__ __forceinline__ v16h load_frag16(const _Float16* tile, int stride,
                                            int row0, int k0, int lane) {
  int r   = row0 + (lane & 15);
  int khi = ((lane >> 4) & 1) << 3;
  v16h f;
#pragma unroll
  for (int j = 0; j < 16; ++j) {
    int k = ((j < 8) ? j : (j + 8)) + khi + k0;
    f[j] = tile[r * stride + k];
  }
  return f;
}

__global__ __launch_bounds__(128)
void fc_gemm_kernel(const float* __restrict__ x,
                    const float* __restrict__ W1,
                    const float* __restrict__ b1,
                    const float* __restrict__ W2,
                    const float* __restrict__ b2,
                    float* __restrict__ xenc) {
  __shared__ _Float16 As[64][KT + APAD];     // x tile       (f32 -> f16)
  __shared__ _Float16 Bs[128][KT + APAD];    // W1 tile      [n][k]
  __shared__ _Float16 Hs[64][128 + HPAD];    // relu(h) tile [m][k]
  __shared__ _Float16 W2s[64][128 + HPAD];   // W2           [n][k]

  const int tid  = threadIdx.x;
  const int wv   = tid >> 5;       // 0..3
  const int lane = tid & 31;
  const int m0   = blockIdx.x * 64;

  // --- preload W2 (64x128 f32) into LDS as f16: 8192 elems / 128 thr ---
  {
    int r = tid >> 1;                 // 0..63
    int c0 = (tid & 1) * 64;          // 0 or 64
#pragma unroll
    for (int q = 0; q < 16; ++q) {
      const float4 v = *(const float4*)&W2[r * 128 + c0 + q * 4];
      W2s[r][c0 + q * 4 + 0] = (_Float16)v.x;
      W2s[r][c0 + q * 4 + 1] = (_Float16)v.y;
      W2s[r][c0 + q * 4 + 2] = (_Float16)v.z;
      W2s[r][c0 + q * 4 + 3] = (_Float16)v.w;
    }
  }

  v8f acc[8] = {};   // this wave: rows [16*wv,16*wv+16) x all 128 cols

  const int tr = tid >> 3;           // 0..15
  const int tc = (tid & 7) * 4;      // 0,4,...,28

  for (int k0 = 0; k0 < 2048; k0 += KT) {
    // cooperative load: x tile 64x32 (f32->f16)
#pragma unroll
    for (int p = 0; p < 4; ++p) {
      int row = tr + p * 16;
      const float4 v = *(const float4*)&x[(size_t)(m0 + row) * 2048 + k0 + tc];
      As[row][tc + 0] = (_Float16)v.x;
      As[row][tc + 1] = (_Float16)v.y;
      As[row][tc + 2] = (_Float16)v.z;
      As[row][tc + 3] = (_Float16)v.w;
    }
    // cooperative load: W1 tile 128x32 (f32->f16), [n][k]
#pragma unroll
    for (int p = 0; p < 8; ++p) {
      int row = tr + p * 16;
      const float4 v = *(const float4*)&W1[(size_t)row * 2048 + k0 + tc];
      Bs[row][tc + 0] = (_Float16)v.x;
      Bs[row][tc + 1] = (_Float16)v.y;
      Bs[row][tc + 2] = (_Float16)v.z;
      Bs[row][tc + 3] = (_Float16)v.w;
    }
    // prefetch next x tile into cache hierarchy (global_prefetch_b8)
    if (k0 + KT < 2048)
      __builtin_prefetch(&x[(size_t)(m0 + tr) * 2048 + k0 + KT + tc], 0, 1);

    __syncthreads();

    v16h a = load_frag16(&As[0][0], KT + APAD, 16 * wv, 0, lane);
#pragma unroll
    for (int t = 0; t < 8; ++t) {
      v16h b = load_frag16(&Bs[0][0], KT + APAD, 16 * t, 0, lane);
      acc[t] = __builtin_amdgcn_wmma_f32_16x16x32_f16(
          false, a, false, b, (short)0, acc[t], false, false);
    }
    __syncthreads();
  }

  // --- epilogue FC1: bias + relu -> Hs (each wave writes its own 16 rows) ---
  {
    const int mo = (lane >= 16) ? 8 : 0;
    const int nl = lane & 15;
#pragma unroll
    for (int t = 0; t < 8; ++t) {
      int n = 16 * t + nl;
      float bias = b1[n];
#pragma unroll
      for (int r = 0; r < 8; ++r) {
        float v = acc[t][r] + bias;
        v = fmaxf(v, 0.0f);
        Hs[16 * wv + r + mo][n] = (_Float16)v;
      }
    }
  }
  __syncthreads();

  // --- FC2: x_enc[m, 0..63] = Hs[m,:] @ W2s^T, K=128 ---
  v8f d[4] = {};
#pragma unroll
  for (int kk = 0; kk < 128; kk += KT) {
    v16h a = load_frag16(&Hs[0][0], 128 + HPAD, 16 * wv, kk, lane);
#pragma unroll
    for (int t = 0; t < 4; ++t) {
      v16h b = load_frag16(&W2s[0][0], 128 + HPAD, 16 * t, kk, lane);
      d[t] = __builtin_amdgcn_wmma_f32_16x16x32_f16(
          false, a, false, b, (short)0, d[t], false, false);
    }
  }
  {
    const int mo = (lane >= 16) ? 8 : 0;
    const int nl = lane & 15;
#pragma unroll
    for (int t = 0; t < 4; ++t) {
      int n = 16 * t + nl;
      float bias = b2[n];
#pragma unroll
      for (int r = 0; r < 8; ++r) {
        int m = m0 + 16 * wv + r + mo;
        xenc[(size_t)m * 64 + n] = d[t][r] + bias;
      }
    }
  }
}

// ---------------------------------------------------------------------------
// Kernel 2: 8-qubit state-vector sim, one wave32 per sample.
// Amplitude index a[7:0]: a[4:0] = lane, a[7:5] = register slot (8 cplx/lane).
// ---------------------------------------------------------------------------
__global__ __launch_bounds__(256)
void quantum_kernel(const float* __restrict__ xenc,
                    const float* __restrict__ qw,
                    const float* __restrict__ W3,
                    const float* __restrict__ b3,
                    const float* __restrict__ W4,
                    const float* __restrict__ b4,
                    float* __restrict__ out) {
  __shared__ float rotm[8][8];  // per wire: m00r,m00i,m01r,m01i,m10r,m10i,m11r,m11i

  const int tid = threadIdx.x;
  if (tid < 8) {
    float phi = qw[tid * 3 + 0];
    float th  = qw[tid * 3 + 1];
    float om  = qw[tid * 3 + 2];
    float c = cosf(th * 0.5f), s = sinf(th * 0.5f);
    float a0 = -0.5f * (phi + om);   // m00 = e^{i a0} c, m11 = conj
    float a1 =  0.5f * (phi - om);   // m01 = -e^{i a1} s, m10 = e^{-i a1} s
    rotm[tid][0] =  cosf(a0) * c;  rotm[tid][1] =  sinf(a0) * c;
    rotm[tid][2] = -cosf(a1) * s;  rotm[tid][3] = -sinf(a1) * s;
    rotm[tid][4] =  cosf(a1) * s;  rotm[tid][5] = -sinf(a1) * s;
    rotm[tid][6] =  cosf(a0) * c;  rotm[tid][7] = -sinf(a0) * c;
  }
  __syncthreads();

  const int wv   = tid >> 5;
  const int lane = tid & 31;
  const int sample = blockIdx.x * 8 + wv;

  float pr[8], pi[8];
#pragma unroll
  for (int r = 0; r < 8; ++r) { pr[r] = 0.0f; pi[r] = 0.0f; }
  if (lane == 0) pr[0] = 1.0f;   // |0...0>

  for (int ch = 0; ch < 8; ++ch) {
    // ---- AngleEmbedding: RY(theta_w) on wire w (real 2x2) ----
#pragma unroll
    for (int w = 0; w < 8; ++w) {
      float th = xenc[(size_t)sample * 64 + ch * 8 + w];
      float c = cosf(th * 0.5f), s = sinf(th * 0.5f);
      if (w < 5) {
        float t = ((lane >> w) & 1) ? s : -s;
#pragma unroll
        for (int r = 0; r < 8; ++r) {
          float qr = __shfl_xor(pr[r], 1 << w, 32);
          float qi = __shfl_xor(pi[r], 1 << w, 32);
          pr[r] = c * pr[r] + t * qr;
          pi[r] = c * pi[r] + t * qi;
        }
      } else {
        int rb = 1 << (w - 5);
#pragma unroll
        for (int r = 0; r < 8; ++r) {
          if (!(r & rb)) {
            int r1 = r | rb;
            float n0r = c * pr[r] - s * pr[r1], n0i = c * pi[r] - s * pi[r1];
            float n1r = s * pr[r] + c * pr[r1], n1i = s * pi[r] + c * pi[r1];
            pr[r] = n0r; pi[r] = n0i; pr[r1] = n1r; pi[r1] = n1i;
          }
        }
      }
    }
    // ---- StronglyEntangling: Rot(phi,theta,omega) on each wire ----
#pragma unroll
    for (int w = 0; w < 8; ++w) {
      float m00r = rotm[w][0], m00i = rotm[w][1];
      float m01r = rotm[w][2], m01i = rotm[w][3];
      float m10r = rotm[w][4], m10i = rotm[w][5];
      float m11r = rotm[w][6], m11i = rotm[w][7];
      if (w < 5) {
        int bit = (lane >> w) & 1;
        float cor = bit ? m11r : m00r, coi = bit ? m11i : m00i;
        float cqr = bit ? m10r : m01r, cqi = bit ? m10i : m01i;
#pragma unroll
        for (int r = 0; r < 8; ++r) {
          float qr = __shfl_xor(pr[r], 1 << w, 32);
          float qi = __shfl_xor(pi[r], 1 << w, 32);
          float nr = cor * pr[r] - coi * pi[r] + cqr * qr - cqi * qi;
          float ni = cor * pi[r] + coi * pr[r] + cqr * qi + cqi * qr;
          pr[r] = nr; pi[r] = ni;
        }
      } else {
        int rb = 1 << (w - 5);
#pragma unroll
        for (int r = 0; r < 8; ++r) {
          if (!(r & rb)) {
            int r1 = r | rb;
            float ar = pr[r], ai = pi[r], br_ = pr[r1], bi_ = pi[r1];
            pr[r]  = m00r * ar - m00i * ai + m01r * br_ - m01i * bi_;
            pi[r]  = m00r * ai + m00i * ar + m01r * bi_ + m01i * br_;
            pr[r1] = m10r * ar - m10i * ai + m11r * br_ - m11i * bi_;
            pi[r1] = m10r * ai + m10i * ar + m11r * bi_ + m11i * br_;
          }
        }
      }
    }
    // ---- CNOT ring, r=1: control w -> target (w+1)%8 ----
#pragma unroll
    for (int w = 0; w < 8; ++w) {
      int ctl = w, tgt = (w + 1) & 7;
      if (ctl < 5 && tgt < 5) {               // both lane bits
        bool sel = (lane >> ctl) & 1;
#pragma unroll
        for (int r = 0; r < 8; ++r) {
          float vr = __shfl_xor(pr[r], 1 << tgt, 32);
          float vi = __shfl_xor(pi[r], 1 << tgt, 32);
          pr[r] = sel ? vr : pr[r];
          pi[r] = sel ? vi : pi[r];
        }
      } else if (ctl < 5) {                   // lane ctrl, reg target
        bool sel = (lane >> ctl) & 1;
        int tb = 1 << (tgt - 5);
#pragma unroll
        for (int r = 0; r < 8; ++r) {
          if (!(r & tb)) {
            int r1 = r | tb;
            float ar = pr[r], ai = pi[r];
            pr[r]  = sel ? pr[r1] : pr[r];   pi[r]  = sel ? pi[r1] : pi[r];
            pr[r1] = sel ? ar     : pr[r1];  pi[r1] = sel ? ai     : pi[r1];
          }
        }
      } else if (tgt < 5) {                   // reg ctrl, lane target
        int cb = 1 << (ctl - 5);
#pragma unroll
        for (int r = 0; r < 8; ++r) {
          float vr = __shfl_xor(pr[r], 1 << tgt, 32);
          float vi = __shfl_xor(pi[r], 1 << tgt, 32);
          if (r & cb) { pr[r] = vr; pi[r] = vi; }
        }
      } else {                                // both reg bits
        int cb = 1 << (ctl - 5), tb = 1 << (tgt - 5);
#pragma unroll
        for (int r = 0; r < 8; ++r) {
          if ((r & cb) && !(r & tb)) {
            int r1 = r | tb;
            float ar = pr[r], ai = pi[r];
            pr[r] = pr[r1]; pi[r] = pi[r1];
            pr[r1] = ar;    pi[r1] = ai;
          }
        }
      }
    }
  }

  // ---- <Z_i> for wires 0..3 (all lane bits) ----
  float amp2 = 0.0f;
#pragma unroll
  for (int r = 0; r < 8; ++r) amp2 += pr[r] * pr[r] + pi[r] * pi[r];

  float z[4];
#pragma unroll
  for (int i = 0; i < 4; ++i) {
    float v = ((lane >> i) & 1) ? -amp2 : amp2;
#pragma unroll
    for (int off = 16; off > 0; off >>= 1) v += __shfl_xor(v, off, 32);
    z[i] = v;
  }

  // ---- tail MLP: relu(z @ W3^T + b3) @ W4^T + b4 ----
  if (lane == 0) {
    float h2[16];
#pragma unroll
    for (int j = 0; j < 16; ++j) {
      float s = b3[j];
#pragma unroll
      for (int i = 0; i < 4; ++i) s += z[i] * W3[j * 4 + i];
      h2[j] = fmaxf(s, 0.0f);
    }
#pragma unroll
    for (int k = 0; k < 3; ++k) {
      float s = b4[k];
#pragma unroll
      for (int j = 0; j < 16; ++j) s += h2[j] * W4[k * 16 + j];
      out[(size_t)sample * 3 + k] = s;
    }
  }
}

// ---------------------------------------------------------------------------
extern "C" void kernel_launch(void* const* d_in, const int* in_sizes, int n_in,
                              void* d_out, int out_size, void* d_ws, size_t ws_size,
                              hipStream_t stream) {
  const float* x  = (const float*)d_in[0];
  const float* W1 = (const float*)d_in[1];
  const float* b1 = (const float*)d_in[2];
  const float* W2 = (const float*)d_in[3];
  const float* b2 = (const float*)d_in[4];
  const float* qw = (const float*)d_in[5];
  const float* W3 = (const float*)d_in[6];
  const float* b3 = (const float*)d_in[7];
  const float* W4 = (const float*)d_in[8];
  const float* b4 = (const float*)d_in[9];
  float* out  = (float*)d_out;
  float* xenc = (float*)d_ws;               // B*64 f32 = 1 MB scratch

  const int B = in_sizes[0] / 2048;         // 4096

  fc_gemm_kernel<<<dim3(B / 64), dim3(128), 0, stream>>>(x, W1, b1, W2, b2, xenc);
  quantum_kernel<<<dim3(B / 8), dim3(256), 0, stream>>>(xenc, qw, W3, b3, W4, b4, out);
}
